// LSTMBased_30812095381744
// MI455X (gfx1250) — matrix operand order repaired and checked by reading, not compiled, not run on hardware
//
#include <hip/hip_runtime.h>
#include <hip/hip_bf16.h>

typedef __attribute__((ext_vector_type(16))) __bf16 bf16x16;
typedef __attribute__((ext_vector_type(8)))  __bf16 bf16x8;
typedef __attribute__((ext_vector_type(8)))  float  f32x8;

#define DEVINL __device__ __forceinline__

DEVINL f32x8 wmma_bf16(bf16x16 a, bf16x16 b, f32x8 c) {
  return __builtin_amdgcn_wmma_f32_16x16x32_bf16(false, a, false, b, (short)0, c, false, false);
}

// A-fragment, row-major A [16 x lda]. lanes 0-15 = rows; elems 0..7 -> K=k0+g*8+i,
// elems 8..15 -> K=k0+16+g*8+i (g = lane>>4).
DEVINL bf16x16 frag_a(const __bf16* A, int lda, int k0) {
  int l = threadIdx.x & 31;
  const __bf16* p = A + (size_t)(l & 15) * lda + k0 + ((l >> 4) << 3);
  bf16x8 lo = *(const bf16x8*)p;
  bf16x8 hi = *(const bf16x8*)(p + 16);
  bf16x16 a;
#pragma unroll
  for (int i = 0; i < 8; ++i) { a[i] = lo[i]; a[8 + i] = hi[i]; }
  return a;
}

// B-fragment for B = W^T, W row-major [N x ldw]: lane l holds col n=n0+(l&15),
// elems 0..7 -> K=k0+g*8+i, elems 8..15 -> K=k0+16+g*8+i (g = lane>>4).
DEVINL bf16x16 frag_bT(const __bf16* W, size_t ldw, int n0, int k0) {
  int l = threadIdx.x & 31;
  const __bf16* p = W + (size_t)(n0 + (l & 15)) * ldw + k0 + ((l >> 4) << 3);
  bf16x8 lo = *(const bf16x8*)p;
  bf16x8 hi = *(const bf16x8*)(p + 16);
  bf16x16 b;
#pragma unroll
  for (int i = 0; i < 8; ++i) { b[i] = lo[i]; b[8 + i] = hi[i]; }
  return b;
}

DEVINL float sigmoidf_(float x) { return 1.0f / (1.0f + __expf(-x)); }

// Device-scope sense-free barrier: counter increases monotonically; target = NWG * phase.
DEVINL void gbar(unsigned* ctr, unsigned target) {
  __threadfence();
  __syncthreads();
  if (threadIdx.x == 0) {
    __hip_atomic_fetch_add(ctr, 1u, __ATOMIC_RELEASE, __HIP_MEMORY_SCOPE_AGENT);
    while (__hip_atomic_load(ctr, __ATOMIC_ACQUIRE, __HIP_MEMORY_SCOPE_AGENT) < target)
      __builtin_amdgcn_s_sleep(1);
  }
  __syncthreads();
  __threadfence();
}

// ---------------------------------------------------------------- utilities
__global__ void k_convert(const float* __restrict__ s, __bf16* __restrict__ d, int n) {
  int i = blockIdx.x * blockDim.x + threadIdx.x;
  if (i < n) d[i] = (__bf16)s[i];
}
__global__ void k_zero(unsigned* __restrict__ p, int n) {
  int i = blockIdx.x * blockDim.x + threadIdx.x;
  if (i < n) p[i] = 0u;
}
__global__ void k_embed_enc(const int* __restrict__ x, const float* __restrict__ emb,
                            __bf16* __restrict__ xs) {
  int bid = blockIdx.x;
  int s = bid >> 4, b = bid & 15;
  int tok = x[b * 256 + s];
  xs[(size_t)bid * 256 + threadIdx.x] = (__bf16)emb[(size_t)tok * 256 + threadIdx.x];
}
__global__ void k_embed_dec(const int* __restrict__ y, const float* __restrict__ emb,
                            __bf16* __restrict__ de) {
  int bid = blockIdx.x;
  int t = bid >> 4, b = bid & 15;
  int tok = (t == 0) ? 1 : y[b * 256 + t];
  de[(size_t)bid * 256 + threadIdx.x] = (__bf16)emb[(size_t)tok * 256 + threadIdx.x];
}

// ------------------------------------------- input GEMM: C_f32 = A*W^T + b1 + b2
// grid: (N/128, M/16), block 256 (8 waves, 1 tile each)
__global__ void k_gemm_xin(const __bf16* __restrict__ A, int lda,
                           const __bf16* __restrict__ W, int ldw,
                           const float* __restrict__ b1, const float* __restrict__ b2,
                           float* __restrict__ C, int N, int K) {
  int wave = threadIdx.x >> 5, lane = threadIdx.x & 31;
  int mt = blockIdx.y;
  int n0 = (blockIdx.x * 8 + wave) * 16;
  int mrow = (lane >> 4) * 8, nc = lane & 15;
  const __bf16* Ab = A + (size_t)mt * 16 * lda;
  float bias = b1[n0 + nc] + b2[n0 + nc];
  f32x8 acc;
#pragma unroll
  for (int j = 0; j < 8; ++j) acc[j] = bias;
  for (int kc = 0; kc < K; kc += 32)
    acc = wmma_bf16(frag_a(Ab, lda, kc), frag_bT(W, (size_t)ldw, n0, kc), acc);
  float* Cb = C + (size_t)mt * 16 * N;
#pragma unroll
  for (int j = 0; j < 8; ++j) Cb[(size_t)(mrow + j) * N + n0 + nc] = acc[j];
}

// ------------------------------------------- encoder LSTM scans, 8 WGs (4 per direction)
// grid = 8 x 512 threads. dir = bid>>2 (0 fwd, 1 bwd, bwd writes cols 512..1023).
// WG owns h columns [sub*128, sub*128+128). h double-buffered in global (bf16).
__global__ void k_lstm_scan8(const float* __restrict__ XinF, const float* __restrict__ XinB,
                             const __bf16* __restrict__ WhF, const __bf16* __restrict__ WhB,
                             __bf16* __restrict__ out1, __bf16* __restrict__ out2,
                             __bf16* __restrict__ hgF, __bf16* __restrict__ hgB,
                             unsigned* __restrict__ bar) {
  const int H = 512, G = 2048, S = 256;
  __shared__ float gjoin[8][4][256];   // [h-tile][gate][m*16+n]  32KB
  __shared__ float cl[16 * 128];       // cell state for owned columns  8KB
  int dir = blockIdx.x >> 2, sub = blockIdx.x & 3;
  const float* Xin = dir ? XinB : XinF;
  const __bf16* Wh = dir ? WhB : WhF;
  __bf16* hg = dir ? hgB : hgF;        // [2][16][512] ping-pong
  int off = dir ? 512 : 0;
  int tid = threadIdx.x, wave = tid >> 5, lane = tid & 31;
  int mrow = (lane >> 4) * 8, nc = lane & 15;
  for (int i = tid; i < 16 * 128; i += 512) cl[i] = 0.0f;
  __syncthreads();
#pragma unroll 1
  for (int step = 0; step < S; ++step) {
    int s = dir ? (S - 1 - step) : step;
    const __bf16* hc = hg + (size_t)(step & 1) * 16 * H;
    __bf16* hn = hg + (size_t)((step + 1) & 1) * 16 * H;
    const float* Xs = Xin + (size_t)s * 16 * G;
    if (step + 1 < S) {   // prefetch next step's gate-bias slab (128KB across 512 threads)
      const char* nx = (const char*)(Xin + (size_t)(dir ? S - 2 - step : step + 1) * 16 * G);
      __builtin_prefetch(nx + tid * 256, 0, 1);
    }
    // gate tiles: wave w -> h-tile ht = w>>1, gates (w&1)*2 + {0,1}
    int ht = wave >> 1;
    int hcol0 = sub * 128 + ht * 16;
#pragma unroll 1
    for (int p = 0; p < 2; ++p) {
      int gt = (wave & 1) * 2 + p;
      int n0 = gt * H + hcol0;
      f32x8 acc;
#pragma unroll
      for (int j = 0; j < 8; ++j) acc[j] = Xs[(size_t)(mrow + j) * G + n0 + nc];
      for (int kc = 0; kc < H; kc += 32)
        acc = wmma_bf16(frag_a(hc, H, kc), frag_bT(Wh, H, n0, kc), acc);
#pragma unroll
      for (int j = 0; j < 8; ++j) gjoin[ht][gt][(mrow + j) * 16 + nc] = acc[j];
    }
    __syncthreads();
    // cell update for owned 16x128 block
    for (int idx = tid; idx < 2048; idx += 512) {
      int m = idx >> 7, clp = idx & 127;
      int tt = clp >> 4, n = clp & 15, cell = m * 16 + n;
      float ig = sigmoidf_(gjoin[tt][0][cell]);
      float fg = sigmoidf_(gjoin[tt][1][cell]);
      float gg = tanhf(gjoin[tt][2][cell]);
      float og = sigmoidf_(gjoin[tt][3][cell]);
      float cc = fg * cl[idx] + ig * gg;
      cl[idx] = cc;
      float h = og * tanhf(cc);
      int col = sub * 128 + clp;
      __bf16 hb = (__bf16)h;
      hn[m * H + col] = hb;
      out1[((size_t)s * 16 + m) * 1024 + off + col] = hb;
      if (out2) out2[((size_t)m * 1024 + off + col) * 256 + s] = hb;
    }
    gbar(bar, 8u * (step + 1));   // h_{t+1} published; safe to read next step
  }
}

// ------------------------------------------- enc_proj[s*16+b][u] = enc . aW1[u]
__global__ void k_encproj(const __bf16* __restrict__ encSB, const float* __restrict__ aW1,
                          float* __restrict__ ep) {
  int idx = blockIdx.x * blockDim.x + threadIdx.x;
  if (idx >= 4096 * 10) return;
  int r = idx / 10, u = idx % 10;
  float acc = 0.0f;
  for (int d = 0; d < 1024; ++d)
    acc += (float)encSB[(size_t)r * 1024 + d] * aW1[u * 1024 + d];
  ep[idx] = acc;
}

// ------------------------------------------- attention + decoder LSTM, 16 WGs x 512
// WG g owns h columns [g*64, g*64+64). Attention (a2/scores/softmax/aj) recomputed
// redundantly per WG (cheap) so only 2 device barriers per step are needed.
__global__ void k_decoder16(const float* __restrict__ XinD,     // [255,16,4096]
                            const __bf16* __restrict__ dwiC,    // d_wi + 256, ldw=1280
                            const __bf16* __restrict__ dwh,     // [4096,1024]
                            const float* __restrict__ encproj,  // [s*16+b][10]
                            const float* __restrict__ aW2, const float* __restrict__ aV,
                            const __bf16* __restrict__ encBDS,  // [16][1024][256]
                            __bf16* __restrict__ hgD,           // [2][16][1024] ping-pong
                            __bf16* __restrict__ ctxg,          // [16][1024]
                            unsigned* __restrict__ bar,
                            __bf16* __restrict__ Hs) {          // [255,16,1024]
  const int HD = 1024;
  __shared__ float a2l[16 * 10];
  __shared__ float scoresl[16 * 256];       // 16KB
  __shared__ __bf16 ajl[16 * 256];          // 8KB
  __shared__ float ctxpart[4][4][256];      // [kq][ntl][m*16+n] 16KB
  __shared__ float gjoin[4][4][256];        // [h-tile][gate][m*16+n] 16KB
  __shared__ float cl[16 * 64];             // 4KB
  int g = blockIdx.x;
  int tid = threadIdx.x, wave = tid >> 5, lane = tid & 31;
  int mrow = (lane >> 4) * 8, nc = lane & 15;
  for (int i = tid; i < 16 * 64; i += 512) cl[i] = 0.0f;
  __syncthreads();
#pragma unroll 1
  for (int t = 0; t < 255; ++t) {
    const __bf16* hc = hgD + (size_t)(t & 1) * 16 * HD;
    __bf16* hn = hgD + (size_t)((t + 1) & 1) * 16 * HD;
    const float* Xt = XinD + (size_t)t * 16 * 4096;
    if (t + 1 < 255) {  // prefetch next step's precomputed gate bias (256KB)
      const char* nx = (const char*)(Xt + 16 * 4096);
      __builtin_prefetch(nx + tid * 256, 0, 1);
      __builtin_prefetch(nx + 131072 + tid * 256, 0, 1);
    }
    // ---- a2[b][u] = h . aW2[u]  (wave w handles batch row b = w)
    {
      int b = wave;
#pragma unroll 1
      for (int u = 0; u < 10; ++u) {
        float acc = 0.0f;
        for (int d = lane; d < 1024; d += 32)
          acc += (float)hc[b * HD + d] * aW2[u * 1024 + d];
#pragma unroll
        for (int o = 16; o > 0; o >>= 1) acc += __shfl_xor(acc, o, 32);
        if (lane == 0) a2l[b * 10 + u] = acc;
      }
    }
    __syncthreads();
    // ---- scores[b][sp]
    for (int idx = tid; idx < 16 * 256; idx += 512) {
      int b = idx >> 8, sp = idx & 255;
      const float* ep = encproj + (sp * 16 + b) * 10;
      float sc = 0.0f;
#pragma unroll
      for (int u = 0; u < 10; ++u) sc += aV[u] * tanhf(ep[u] + a2l[b * 10 + u]);
      scoresl[b * 256 + sp] = sc;
    }
    __syncthreads();
    // ---- softmax rows (wave b handles row b)
    {
      int b = wave;
      float mx = -1e30f;
      for (int sp = lane; sp < 256; sp += 32) mx = fmaxf(mx, scoresl[b * 256 + sp]);
#pragma unroll
      for (int o = 16; o > 0; o >>= 1) mx = fmaxf(mx, __shfl_xor(mx, o, 32));
      float sum = 0.0f;
      for (int sp = lane; sp < 256; sp += 32) {
        float e = __expf(scoresl[b * 256 + sp] - mx);
        scoresl[b * 256 + sp] = e;
        sum += e;
      }
#pragma unroll
      for (int o = 16; o > 0; o >>= 1) sum += __shfl_xor(sum, o, 32);
      float inv = 1.0f / sum;
      for (int sp = lane; sp < 256; sp += 32)
        ajl[b * 256 + sp] = (__bf16)(scoresl[b * 256 + sp] * inv);
    }
    __syncthreads();
    // ---- ctx slice: blockdiag(aj)[16 x 4096] @ E[4096 x 1024]; WG g owns n-tiles g*4..g*4+3
    // wave w: n-tile local ntl = w&3, K-quarter kq = w>>2 (32 of 128 chunks)
    {
      int ntl = wave & 3, kq = wave >> 2;
      int n0 = (g * 4 + ntl) * 16;
      f32x8 acc;
#pragma unroll
      for (int j = 0; j < 8; ++j) acc[j] = 0.0f;
#pragma unroll 1
      for (int cc = 0; cc < 32; ++cc) {
        int kc = kq * 32 + cc;                 // chunk of 32 K-rows, all from batch bb
        int bb = kc >> 3, soff = (kc & 7) * 32;
        bf16x16 a;
        int m = lane & 15, gg = lane >> 4;
        if (m == bb) {
          const __bf16* p = ajl + bb * 256 + soff + (gg << 3);
          bf16x8 lo = *(const bf16x8*)p;
          bf16x8 hi = *(const bf16x8*)(p + 16);
#pragma unroll
          for (int i = 0; i < 8; ++i) { a[i] = lo[i]; a[8 + i] = hi[i]; }
        } else {
#pragma unroll
          for (int i = 0; i < 16; ++i) a[i] = (__bf16)0.0f;
        }
        acc = wmma_bf16(a, frag_bT(encBDS + (size_t)bb * 1024 * 256, 256, n0, soff), acc);
      }
#pragma unroll
      for (int j = 0; j < 8; ++j) ctxpart[kq][ntl][(mrow + j) * 16 + nc] = acc[j];
    }
    __syncthreads();
    for (int idx = tid; idx < 1024; idx += 512) {   // reduce 4 K-partials, publish ctx
      int ntl = idx >> 8, cell = idx & 255;
      float s = ctxpart[0][ntl][cell] + ctxpart[1][ntl][cell] +
                ctxpart[2][ntl][cell] + ctxpart[3][ntl][cell];
      ctxg[(cell >> 4) * HD + (g * 4 + ntl) * 16 + (cell & 15)] = (__bf16)s;
    }
    gbar(bar, 16u * (2 * t + 1));     // full ctx visible to all WGs
    // ---- gates: wave w -> h-tile ht = w>>2 (of WG's 4), gate gt = w&3
    {
      int ht = wave >> 2, gt = wave & 3;
      int n0 = gt * HD + (g * 4 + ht) * 16;
      f32x8 acc;
#pragma unroll
      for (int j = 0; j < 8; ++j) acc[j] = Xt[(size_t)(mrow + j) * 4096 + n0 + nc];
      for (int kc = 0; kc < HD; kc += 32)
        acc = wmma_bf16(frag_a(ctxg, HD, kc), frag_bT(dwiC, 1280, n0, kc), acc);
      for (int kc = 0; kc < HD; kc += 32)
        acc = wmma_bf16(frag_a(hc, HD, kc), frag_bT(dwh, HD, n0, kc), acc);
#pragma unroll
      for (int j = 0; j < 8; ++j) gjoin[ht][gt][(mrow + j) * 16 + nc] = acc[j];
    }
    __syncthreads();
    // ---- cell update for owned 16x64 block
    for (int idx = tid; idx < 1024; idx += 512) {
      int ht = idx >> 8, cell = idx & 255;
      int m = cell >> 4, n = cell & 15;
      int clp = ht * 16 + n;
      float ig = sigmoidf_(gjoin[ht][0][cell]);
      float fg = sigmoidf_(gjoin[ht][1][cell]);
      float gg = tanhf(gjoin[ht][2][cell]);
      float og = sigmoidf_(gjoin[ht][3][cell]);
      float ccv = fg * cl[m * 64 + clp] + ig * gg;
      cl[m * 64 + clp] = ccv;
      float h = og * tanhf(ccv);
      int col = g * 64 + clp;
      __bf16 hb = (__bf16)h;
      hn[m * HD + col] = hb;
      Hs[((size_t)t * 16 + m) * HD + col] = hb;
    }
    gbar(bar, 16u * (2 * t + 2));     // h_{t+1} published
  }
}

// ------------------------------------------- vocab GEMM (register-blocked 3M x 2N)
// grid (16384/256, 255/3), block 256 (8 waves); wave: 2 n-tiles, 3 m-tiles.
__global__ void k_vocab(const __bf16* __restrict__ Hs, const __bf16* __restrict__ vW,
                        const float* __restrict__ vb, float* __restrict__ out) {
  int wave = threadIdx.x >> 5, lane = threadIdx.x & 31;
  int mt0 = blockIdx.y * 3;                       // m-tile == decoder step t
  int n0 = (blockIdx.x * 16 + wave * 2) * 16;     // first of two n-tiles
  int mrow = (lane >> 4) * 8, nc = lane & 15;
  f32x8 acc[3][2];
#pragma unroll
  for (int i = 0; i < 3; ++i) {
    for (int j = 0; j < 8; ++j) { acc[i][0][j] = vb[n0 + nc]; acc[i][1][j] = vb[n0 + 16 + nc]; }
  }
  for (int kc = 0; kc < 1024; kc += 32) {
    bf16x16 b0 = frag_bT(vW, 1024, n0, kc);
    bf16x16 b1 = frag_bT(vW, 1024, n0 + 16, kc);
#pragma unroll
    for (int i = 0; i < 3; ++i) {
      bf16x16 a = frag_a(Hs + (size_t)(mt0 + i) * 16 * 1024, 1024, kc);
      acc[i][0] = wmma_bf16(a, b0, acc[i][0]);
      acc[i][1] = wmma_bf16(a, b1, acc[i][1]);
    }
  }
#pragma unroll
  for (int i = 0; i < 3; ++i) {
    int t = mt0 + i;
#pragma unroll
    for (int j = 0; j < 8; ++j) {
      int b = mrow + j;
      out[((size_t)b * 255 + t) * 16384 + n0 + nc] = acc[i][0][j];
      out[((size_t)b * 255 + t) * 16384 + n0 + 16 + nc] = acc[i][1][j];
    }
  }
}

// ================================================================= host
extern "C" void kernel_launch(void* const* d_in, const int* in_sizes, int n_in,
                              void* d_out, int out_size, void* d_ws, size_t ws_size,
                              hipStream_t stream) {
  (void)in_sizes; (void)n_in; (void)out_size; (void)ws_size;
  const int*   x       = (const int*)d_in[0];
  const int*   y       = (const int*)d_in[1];
  const float* emb_enc = (const float*)d_in[2];
  const float* emb_dec = (const float*)d_in[3];
  const float* e0f_wi = (const float*)d_in[4],  *e0f_wh = (const float*)d_in[5];
  const float* e0f_bi = (const float*)d_in[6],  *e0f_bh = (const float*)d_in[7];
  const float* e0b_wi = (const float*)d_in[8],  *e0b_wh = (const float*)d_in[9];
  const float* e0b_bi = (const float*)d_in[10], *e0b_bh = (const float*)d_in[11];
  const float* e1f_wi = (const float*)d_in[12], *e1f_wh = (const float*)d_in[13];
  const float* e1f_bi = (const float*)d_in[14], *e1f_bh = (const float*)d_in[15];
  const float* e1b_wi = (const float*)d_in[16], *e1b_wh = (const float*)d_in[17];
  const float* e1b_bi = (const float*)d_in[18], *e1b_bh = (const float*)d_in[19];
  const float* d_wi = (const float*)d_in[20], *d_wh = (const float*)d_in[21];
  const float* d_bi = (const float*)d_in[22], *d_bh = (const float*)d_in[23];
  const float* aW1 = (const float*)d_in[24], *aW2 = (const float*)d_in[25];
  const float* aV  = (const float*)d_in[26];
  const float* vW  = (const float*)d_in[27], *vb = (const float*)d_in[28];
  float* out = (float*)d_out;
  char* ws = (char*)d_ws;

  size_t off = 0;
  auto take = [&](size_t bytes) { size_t o = off; off += (bytes + 255) & ~(size_t)255; return o; };
  size_t o_e0f_wi = take((size_t)2048 * 256 * 2);
  size_t o_e0f_wh = take((size_t)2048 * 512 * 2);
  size_t o_e0b_wi = take((size_t)2048 * 256 * 2);
  size_t o_e0b_wh = take((size_t)2048 * 512 * 2);
  size_t o_e1f_wi = take((size_t)2048 * 1024 * 2);
  size_t o_e1f_wh = take((size_t)2048 * 512 * 2);
  size_t o_e1b_wi = take((size_t)2048 * 1024 * 2);
  size_t o_e1b_wh = take((size_t)2048 * 512 * 2);
  size_t o_dwi    = take((size_t)4096 * 1280 * 2);
  size_t o_dwh    = take((size_t)4096 * 1024 * 2);
  size_t o_vw     = take((size_t)16384 * 1024 * 2);
  size_t o_xs     = take((size_t)256 * 16 * 256 * 2);
  size_t o_de     = take((size_t)255 * 16 * 256 * 2);
  size_t o_l0     = take((size_t)256 * 16 * 1024 * 2);
  size_t o_encSB  = take((size_t)256 * 16 * 1024 * 2);
  size_t o_encBDS = take((size_t)16 * 1024 * 256 * 2);
  size_t o_ep     = take((size_t)4096 * 10 * 4);
  size_t o_hs     = take((size_t)255 * 16 * 1024 * 2);
  // cross-WG sync state (zeroed before each persistent launch)
  size_t o_hgF  = take((size_t)2 * 16 * 512 * 2);
  size_t o_hgB  = take((size_t)2 * 16 * 512 * 2);
  size_t o_hgD  = take((size_t)2 * 16 * 1024 * 2);
  size_t o_ctxg = take((size_t)16 * 1024 * 2);
  size_t o_bar  = take((size_t)256);
  size_t o_arena = take((size_t)67108864);   // Xin0f/b or Xin1f/b halves, or XinD
  int n_sync = (int)((o_bar + 256 - o_hgF) / 4);

  auto bw = [&](size_t o) { return (__bf16*)(ws + o); };
  auto cvt = [&](const float* src, size_t dst, size_t n) {
    k_convert<<<dim3((unsigned)((n + 255) / 256)), 256, 0, stream>>>(src, bw(dst), (int)n);
  };
  auto zero_sync = [&]() {
    k_zero<<<dim3((unsigned)((n_sync + 255) / 256)), 256, 0, stream>>>(
        (unsigned*)(ws + o_hgF), n_sync);
  };
  cvt(e0f_wi, o_e0f_wi, (size_t)2048 * 256);
  cvt(e0f_wh, o_e0f_wh, (size_t)2048 * 512);
  cvt(e0b_wi, o_e0b_wi, (size_t)2048 * 256);
  cvt(e0b_wh, o_e0b_wh, (size_t)2048 * 512);
  cvt(e1f_wi, o_e1f_wi, (size_t)2048 * 1024);
  cvt(e1f_wh, o_e1f_wh, (size_t)2048 * 512);
  cvt(e1b_wi, o_e1b_wi, (size_t)2048 * 1024);
  cvt(e1b_wh, o_e1b_wh, (size_t)2048 * 512);
  cvt(d_wi,   o_dwi,    (size_t)4096 * 1280);
  cvt(d_wh,   o_dwh,    (size_t)4096 * 1024);
  cvt(vW,     o_vw,     (size_t)16384 * 1024);

  k_embed_enc<<<4096, 256, 0, stream>>>(x, emb_enc, bw(o_xs));
  k_embed_dec<<<255 * 16, 256, 0, stream>>>(y, emb_dec, bw(o_de));

  float* Xin0f = (float*)(ws + o_arena);
  float* Xin0b = (float*)(ws + o_arena + 33554432);
  float* XinD  = (float*)(ws + o_arena);
  unsigned* bar = (unsigned*)(ws + o_bar);

  // layer 0
  k_gemm_xin<<<dim3(16, 256), 256, 0, stream>>>(bw(o_xs), 256, bw(o_e0f_wi), 256,
                                                e0f_bi, e0f_bh, Xin0f, 2048, 256);
  k_gemm_xin<<<dim3(16, 256), 256, 0, stream>>>(bw(o_xs), 256, bw(o_e0b_wi), 256,
                                                e0b_bi, e0b_bh, Xin0b, 2048, 256);
  zero_sync();
  k_lstm_scan8<<<8, 512, 0, stream>>>(Xin0f, Xin0b, bw(o_e0f_wh), bw(o_e0b_wh),
                                      bw(o_l0), (__bf16*)nullptr, bw(o_hgF), bw(o_hgB), bar);
  // layer 1
  k_gemm_xin<<<dim3(16, 256), 256, 0, stream>>>(bw(o_l0), 1024, bw(o_e1f_wi), 1024,
                                                e1f_bi, e1f_bh, Xin0f, 2048, 1024);
  k_gemm_xin<<<dim3(16, 256), 256, 0, stream>>>(bw(o_l0), 1024, bw(o_e1b_wi), 1024,
                                                e1b_bi, e1b_bh, Xin0b, 2048, 1024);
  zero_sync();
  k_lstm_scan8<<<8, 512, 0, stream>>>(Xin0f, Xin0b, bw(o_e1f_wh), bw(o_e1b_wh),
                                      bw(o_encSB), bw(o_encBDS), bw(o_hgF), bw(o_hgB), bar);
  // attention prep + decoder
  k_encproj<<<160, 256, 0, stream>>>(bw(o_encSB), aW1, (float*)(ws + o_ep));
  k_gemm_xin<<<dim3(32, 255), 256, 0, stream>>>(bw(o_de), 256, bw(o_dwi), 1280,
                                                d_bi, d_bh, XinD, 4096, 256);
  zero_sync();
  k_decoder16<<<16, 512, 0, stream>>>(XinD, bw(o_dwi) + 256, bw(o_dwh),
                                      (const float*)(ws + o_ep), aW2, aV,
                                      bw(o_encBDS), bw(o_hgD), bw(o_ctxg), bar, bw(o_hs));
  // vocab projection
  k_vocab<<<dim3(64, 85), 256, 0, stream>>>(bw(o_hs), bw(o_vw), vb, out);
}
